// Transformer_64132451664610
// MI455X (gfx1250) — matrix operand order repaired and checked by reading, not compiled, not run on hardware
//
#include <hip/hip_runtime.h>
#include <hip/hip_bf16.h>
#include <math.h>

typedef __attribute__((ext_vector_type(16))) __bf16 v16bf;
typedef __attribute__((ext_vector_type(8)))  float  v8f;

union UF { unsigned int u[8]; v16bf v; };

__device__ __forceinline__ unsigned short f2bf(float f) {
    unsigned int u = __float_as_uint(f);
    return (unsigned short)((u + 0x7FFFu + ((u >> 16) & 1u)) >> 16);  // RNE
}
// ISA 7.12.2 16-bit A-matrix 16x32: VGPR j holds K pair base:
__device__ __forceinline__ int kA(int j, int hl) { return ((j & 3) << 1) + ((j >> 2) << 4) + (hl << 3); }
// B-matrix 32x16: same K order as A (VGPR j pair base):
__device__ __forceinline__ int kB(int j, int hl) { return ((j & 3) << 1) + ((j >> 2) << 4) + (hl << 3); }

// Issue two async 16B global->LDS copies (32B per thread), ASYNCcnt += 2.
#define ASYNC_CP32B(ldsoff, gsrc)                                              \
    asm volatile("global_load_async_to_lds_b128 %0, %1, off\n\t"               \
                 "global_load_async_to_lds_b128 %0, %1, off offset:16"         \
                 :: "v"(ldsoff), "v"(gsrc) : "memory")
#define WAIT_ASYNC4()  asm volatile("s_wait_asynccnt 0x4" ::: "memory")
#define WAIT_ASYNC0()  asm volatile("s_wait_asynccnt 0x0" ::: "memory")

// ------------------------------------------------- weight transpose ------
// src: f32 [R,C] row-major  ->  dst: bf16 [C,R] row-major (i.e. src^T)
__global__ __launch_bounds__(256) void transpose_bf16_kernel(
    const float* __restrict__ src, unsigned short* __restrict__ dst, int R, int C)
{
    __shared__ unsigned short tile[32][33];
    const int r0 = blockIdx.y * 32, c0 = blockIdx.x * 32;
    const int tr = threadIdx.x >> 3;        // 0..31
    const int tc4 = (threadIdx.x & 7) * 4;  // 0,4,...,28
    float4 v = *(const float4*)(src + (size_t)(r0 + tr) * C + c0 + tc4);
    tile[tr][tc4 + 0] = f2bf(v.x); tile[tr][tc4 + 1] = f2bf(v.y);
    tile[tr][tc4 + 2] = f2bf(v.z); tile[tr][tc4 + 3] = f2bf(v.w);
    __syncthreads();
#pragma unroll
    for (int i = 0; i < 4; i++)
        dst[(size_t)(c0 + tr) * R + r0 + tc4 + i] = tile[tc4 + i][tr];
}

// ---------------------------------------------------------------- GEMM ----
// C[M,N] = epilogue(A[M,K] @ Bt[N,K]^T); bf16 operands, f32 accumulate.
// Double-buffered LDS fed by global_load_async_to_lds_b128; K-loop unrolled
// by two tiles (one per buffer) for a branch-free steady state.
#define BMT 128
#define BNT 128
#define BKT 32
__global__ __launch_bounds__(256) void gemm_bf16_async(
    const unsigned short* __restrict__ A,   // [M,K] bf16
    const unsigned short* __restrict__ Bt,  // [N,K] bf16 (pre-transposed weight)
    float* __restrict__ Cf,                 // optional f32 output
    unsigned short* __restrict__ Cbf,       // optional bf16 output
    const float* __restrict__ bias,
    const float* __restrict__ resid,
    int M, int N, int K, int relu)
{
    __shared__ unsigned short aL[2][BMT * BKT];  // [m][k]
    __shared__ unsigned short bL[2][BNT * BKT];  // [n][k]
    const int t = threadIdx.x;
    const int lane = t & 31, w = t >> 5;
    const int hl = lane >> 4, lp = lane & 15;
    const int bm0 = blockIdx.y * BMT, bn0 = blockIdx.x * BNT;
    const int wm = (w >> 1) * 32, wn = (w & 1) * 64;

    v8f acc[2][4];
#pragma unroll
    for (int i = 0; i < 2; i++)
#pragma unroll
        for (int j = 0; j < 4; j++)
#pragma unroll
            for (int e = 0; e < 8; e++) acc[i][j][e] = 0.0f;

    // per-thread async-copy assignment: row = t>>1, 32B chunk = (t&1)*16 elems
    const int cr = t >> 1, cc = (t & 1) * 16;
    const unsigned short* aSrc = A + (size_t)(bm0 + cr) * K + cc;
    const unsigned short* bSrc = Bt + (size_t)(bn0 + cr) * K + cc;
    const unsigned aOff0 = (unsigned)(uintptr_t)(&aL[0][cr * BKT + cc]);
    const unsigned aOff1 = (unsigned)(uintptr_t)(&aL[1][cr * BKT + cc]);
    const unsigned bOff0 = (unsigned)(uintptr_t)(&bL[0][cr * BKT + cc]);
    const unsigned bOff1 = (unsigned)(uintptr_t)(&bL[1][cr * BKT + cc]);

    const unsigned short* aB0 = aL[0];
    const unsigned short* bB0 = bL[0];
    const unsigned short* aB1 = aL[1];
    const unsigned short* bB1 = bL[1];

    auto compute = [&](const unsigned short* aB, const unsigned short* bB) {
        UF af[2], bf[4];
#pragma unroll
        for (int j = 0; j < 8; j++) {
            const int ka = kA(j, hl), kb = kB(j, hl);
#pragma unroll
            for (int i = 0; i < 2; i++)
                af[i].u[j] = *(const unsigned int*)(aB + (wm + i * 16 + lp) * BKT + ka);
#pragma unroll
            for (int i = 0; i < 4; i++)
                bf[i].u[j] = *(const unsigned int*)(bB + (wn + i * 16 + lp) * BKT + kb);
        }
#pragma unroll
        for (int i = 0; i < 2; i++)
#pragma unroll
            for (int j = 0; j < 4; j++)
                acc[i][j] = __builtin_amdgcn_wmma_f32_16x16x32_bf16(
                    false, af[i].v, false, bf[j].v, (short)0, acc[i][j], false, false);
    };

    // prologue: tiles 0,1 in flight (ASYNCcnt = 8)
    ASYNC_CP32B(aOff0, aSrc);
    ASYNC_CP32B(bOff0, bSrc);
    ASYNC_CP32B(aOff1, aSrc + BKT);
    ASYNC_CP32B(bOff1, bSrc + BKT);
    const unsigned short* aNext = aSrc + 2 * BKT;
    const unsigned short* bNext = bSrc + 2 * BKT;

    // steady state: branch-free two-tile body
    const int pairs = K / (2 * BKT);
    for (int it = 0; it < pairs - 1; ++it) {
        WAIT_ASYNC4();        // tile for buf0 landed
        __syncthreads();
        compute(aB0, bB0);
        __syncthreads();
        ASYNC_CP32B(aOff0, aNext);
        ASYNC_CP32B(bOff0, bNext);
        aNext += BKT; bNext += BKT;

        WAIT_ASYNC4();        // tile for buf1 landed
        __syncthreads();
        compute(aB1, bB1);
        __syncthreads();
        ASYNC_CP32B(aOff1, aNext);
        ASYNC_CP32B(bOff1, bNext);
        aNext += BKT; bNext += BKT;
    }
    // peeled final pair
    WAIT_ASYNC4();
    __syncthreads();
    compute(aB0, bB0);
    __syncthreads();
    WAIT_ASYNC0();
    __syncthreads();
    compute(aB1, bB1);

    // ---- fused epilogue: bias / relu / residual, f32 and/or bf16 out ----
#pragma unroll
    for (int i = 0; i < 2; i++) {
#pragma unroll
        for (int j = 0; j < 4; j++) {
            const int col = bn0 + wn + j * 16 + lp;
            const float bv = bias ? bias[col] : 0.0f;
#pragma unroll
            for (int r = 0; r < 8; r++) {
                const int row = bm0 + wm + i * 16 + r + hl * 8;
                float v = acc[i][j][r] + bv;
                if (relu) v = v > 0.0f ? v : 0.0f;
                if (resid) v += resid[(size_t)row * N + col];
                if (Cf)  Cf[(size_t)row * N + col] = v;
                if (Cbf) Cbf[(size_t)row * N + col] = f2bf(v);
            }
        }
    }
}

// ----------------------------------------------------------- attention ----
// One wave per (b, head, 16-query tile). n=1024, head dim 64. bf16 QKV in,
// bf16 out; f32 scores + softmax staged in 64KB dynamic LDS.
__global__ __launch_bounds__(32) void attention_kernel(
    const unsigned short* __restrict__ Q, const unsigned short* __restrict__ Km,
    const unsigned short* __restrict__ V, unsigned short* __restrict__ O)
{
    extern __shared__ float S[];                         // 16 x 1024 f32
    unsigned short* P = (unsigned short*)S + 16 * 1024;  // bf16 probs (upper 32KB)

    const int lane = threadIdx.x & 31;
    const int hl = lane >> 4, lp = lane & 15;
    const int qt   = blockIdx.x & 63;
    const int head = (blockIdx.x >> 6) & 15;
    const int b    = blockIdx.x >> 10;

    const unsigned short* Qb = Q + (size_t)b * 1024 * 1024 + head * 64;
    const unsigned short* Kb = Km + (size_t)b * 1024 * 1024 + head * 64;
    const unsigned short* Vb = V + (size_t)b * 1024 * 1024 + head * 64;
    unsigned short*       Ob = O + (size_t)b * 1024 * 1024 + head * 64;

    UF qf[2];
#pragma unroll
    for (int j = 0; j < 8; j++) {
        const int ka = kA(j, hl);
#pragma unroll
        for (int kk = 0; kk < 2; kk++)
            qf[kk].u[j] = *(const unsigned int*)(Qb + (size_t)(qt * 16 + lp) * 1024 + kk * 32 + ka);
    }

    // S = Q K^T / 8 over 64 key tiles
    for (int kt = 0; kt < 64; kt++) {
        UF kf[2];
#pragma unroll
        for (int j = 0; j < 8; j++) {
            const int kb = kB(j, hl);
#pragma unroll
            for (int kk = 0; kk < 2; kk++)
                kf[kk].u[j] = *(const unsigned int*)(Kb + (size_t)(kt * 16 + lp) * 1024 + kk * 32 + kb);
        }
        v8f s;
#pragma unroll
        for (int e = 0; e < 8; e++) s[e] = 0.0f;
        s = __builtin_amdgcn_wmma_f32_16x16x32_bf16(false, qf[0].v, false, kf[0].v, (short)0, s, false, false);
        s = __builtin_amdgcn_wmma_f32_16x16x32_bf16(false, qf[1].v, false, kf[1].v, (short)0, s, false, false);
#pragma unroll
        for (int r = 0; r < 8; r++)
            S[(r + hl * 8) * 1024 + kt * 16 + lp] = s[r] * 0.125f;
    }
    __syncthreads();

    // row softmax -> bf16 P (descending rows: safe in-buffer packing)
    for (int row = 15; row >= 0; row--) {
        float vals[32];
        float mx = -1e30f;
#pragma unroll
        for (int i = 0; i < 32; i++) {
            vals[i] = S[row * 1024 + i * 32 + lane];
            mx = fmaxf(mx, vals[i]);
        }
#pragma unroll
        for (int off = 16; off > 0; off >>= 1) mx = fmaxf(mx, __shfl_xor(mx, off, 32));
        float sum = 0.0f;
#pragma unroll
        for (int i = 0; i < 32; i++) { vals[i] = __expf(vals[i] - mx); sum += vals[i]; }
#pragma unroll
        for (int off = 16; off > 0; off >>= 1) sum += __shfl_xor(sum, off, 32);
        const float inv = 1.0f / sum;
#pragma unroll
        for (int i = 0; i < 32; i++)
            P[row * 1024 + i * 32 + lane] = f2bf(vals[i] * inv);
    }
    __syncthreads();

    // O = P @ V : K = 1024 (32 steps), N = 64 (4 tiles)
    v8f acc[4];
#pragma unroll
    for (int j = 0; j < 4; j++)
#pragma unroll
        for (int e = 0; e < 8; e++) acc[j][e] = 0.0f;

    for (int ks = 0; ks < 32; ks++) {
        UF pf, vf[4];
#pragma unroll
        for (int j = 0; j < 8; j++) {
            const int ka = kA(j, hl), kb = kB(j, hl);
            pf.u[j] = *(const unsigned int*)(&P[lp * 1024 + ks * 32 + ka]);
#pragma unroll
            for (int nj = 0; nj < 4; nj++) {
                const unsigned v0 = Vb[(size_t)(ks * 32 + kb + 0) * 1024 + nj * 16 + lp];
                const unsigned v1 = Vb[(size_t)(ks * 32 + kb + 1) * 1024 + nj * 16 + lp];
                vf[nj].u[j] = v0 | (v1 << 16);
            }
        }
#pragma unroll
        for (int nj = 0; nj < 4; nj++)
            acc[nj] = __builtin_amdgcn_wmma_f32_16x16x32_bf16(
                false, pf.v, false, vf[nj].v, (short)0, acc[nj], false, false);
    }
#pragma unroll
    for (int nj = 0; nj < 4; nj++)
#pragma unroll
        for (int r = 0; r < 8; r++)
            Ob[(size_t)(qt * 16 + r + hl * 8) * 1024 + nj * 16 + lp] = f2bf(acc[nj][r]);
}

// ----------------------------------------------------------- layernorm ----
__global__ __launch_bounds__(256) void layernorm_kernel(
    const float* __restrict__ X, const float* __restrict__ g,
    const float* __restrict__ bta, float* __restrict__ Y,
    unsigned short* __restrict__ Ybf)
{
    __shared__ float red[8];
    const int row = blockIdx.x;
    const float* x = X + (size_t)row * 1024;
    const int t = threadIdx.x;

    float v[4];
    float s = 0.0f;
#pragma unroll
    for (int i = 0; i < 4; i++) { v[i] = x[t + 256 * i]; s += v[i]; }
#pragma unroll
    for (int off = 16; off > 0; off >>= 1) s += __shfl_xor(s, off, 32);
    if ((t & 31) == 0) red[t >> 5] = s;
    __syncthreads();
    float tot = 0.0f;
#pragma unroll
    for (int i = 0; i < 8; i++) tot += red[i];
    const float mean = tot * (1.0f / 1024.0f);
    __syncthreads();

    float s2 = 0.0f;
#pragma unroll
    for (int i = 0; i < 4; i++) { const float d = v[i] - mean; s2 += d * d; }
#pragma unroll
    for (int off = 16; off > 0; off >>= 1) s2 += __shfl_xor(s2, off, 32);
    if ((t & 31) == 0) red[t >> 5] = s2;
    __syncthreads();
    float tot2 = 0.0f;
#pragma unroll
    for (int i = 0; i < 8; i++) tot2 += red[i];
    const float inv = rsqrtf(tot2 * (1.0f / 1024.0f) + 1e-5f);

#pragma unroll
    for (int i = 0; i < 4; i++) {
        const int c = t + 256 * i;
        const float yv = (v[i] - mean) * inv * g[c] + bta[c];
        Y[(size_t)row * 1024 + c] = yv;
        if (Ybf) Ybf[(size_t)row * 1024 + c] = f2bf(yv);
    }
}

// ------------------------------------------------- embedding + pos enc ----
__global__ __launch_bounds__(256) void embed_pe_kernel(
    const int* __restrict__ x, const float* __restrict__ emb,
    float* __restrict__ h, unsigned short* __restrict__ hbf)
{
    const int row = blockIdx.x;  // b*1024 + pos
    const int pos = row & 1023;
    const int tok = x[row];
    const int d0 = threadIdx.x * 4;
    const float c = -9.210340371976184f / 1024.0f;  // -ln(10000)/D
#pragma unroll
    for (int dd = 0; dd < 4; dd++) {
        const int d = d0 + dd;
        const float ang = (float)pos * __expf((float)(2 * (d >> 1)) * c);
        const float pe = (d & 1) ? __cosf(ang) : __sinf(ang);
        const float v = emb[(size_t)tok * 1024 + d] + pe;
        h[(size_t)row * 1024 + d] = v;
        hbf[(size_t)row * 1024 + d] = f2bf(v);
    }
}

// ------------------------------------------------------------- launch -----
extern "C" void kernel_launch(void* const* d_in, const int* in_sizes, int n_in,
                              void* d_out, int out_size, void* d_ws, size_t ws_size,
                              hipStream_t stream) {
    const int*   x    = (const int*)d_in[0];
    const float* emb  = (const float*)d_in[1];
    const float* Wq   = (const float*)d_in[2];
    const float* Wk   = (const float*)d_in[3];
    const float* Wv   = (const float*)d_in[4];
    const float* Wc   = (const float*)d_in[5];
    const float* W1   = (const float*)d_in[6];
    const float* b1   = (const float*)d_in[7];
    const float* W2   = (const float*)d_in[8];
    const float* b2   = (const float*)d_in[9];
    const float* g1   = (const float*)d_in[10];
    const float* be1  = (const float*)d_in[11];
    const float* g2   = (const float*)d_in[12];
    const float* be2  = (const float*)d_in[13];
    float* out = (float*)d_out;

    // workspace carve-up (~150MB)
    const size_t T = (size_t)2048 * 1024;  // 2M tokens*D
    char* p = (char*)d_ws;
    float* hbuf = (float*)p;              p += T * 4;
    float* tb   = (float*)p;              p += T * 4;
    unsigned short* hbf = (unsigned short*)p;  p += T * 2;
    unsigned short* qb  = (unsigned short*)p;  p += T * 2;
    unsigned short* kb  = (unsigned short*)p;  p += T * 2;
    unsigned short* vb  = (unsigned short*)p;  p += T * 2;
    unsigned short* abf = (unsigned short*)p;  p += T * 2;
    unsigned short* fbf = (unsigned short*)p;  p += (size_t)2048 * 4096 * 2;
    unsigned short* wT  = (unsigned short*)p;  // 48M bf16: per layer 12M
    const size_t LW = (size_t)12 * 1024 * 1024;

    // ---- per-call weight convert+transpose to bf16 (L2-resident reuse) ----
    for (int l = 0; l < 4; l++) {
        unsigned short* wqT = wT + l * LW;
        unsigned short* wkT = wqT + (size_t)1024 * 1024;
        unsigned short* wvT = wkT + (size_t)1024 * 1024;
        unsigned short* wcT = wvT + (size_t)1024 * 1024;
        unsigned short* w1T = wcT + (size_t)1024 * 1024;
        unsigned short* w2T = w1T + (size_t)1024 * 4096;
        transpose_bf16_kernel<<<dim3(32, 32), 256, 0, stream>>>(Wq + (size_t)l * 1024 * 1024, wqT, 1024, 1024);
        transpose_bf16_kernel<<<dim3(32, 32), 256, 0, stream>>>(Wk + (size_t)l * 1024 * 1024, wkT, 1024, 1024);
        transpose_bf16_kernel<<<dim3(32, 32), 256, 0, stream>>>(Wv + (size_t)l * 1024 * 1024, wvT, 1024, 1024);
        transpose_bf16_kernel<<<dim3(32, 32), 256, 0, stream>>>(Wc + (size_t)l * 1024 * 1024, wcT, 1024, 1024);
        transpose_bf16_kernel<<<dim3(128, 32), 256, 0, stream>>>(W1 + (size_t)l * 1024 * 4096, w1T, 1024, 4096);
        transpose_bf16_kernel<<<dim3(32, 128), 256, 0, stream>>>(W2 + (size_t)l * 4096 * 1024, w2T, 4096, 1024);
    }

    embed_pe_kernel<<<2048, 256, 0, stream>>>(x, emb, hbuf, hbf);

    const dim3 gP(1024 / BNT, 2048 / BMT);
    const dim3 gF(4096 / BNT, 2048 / BMT);
    const size_t attn_lds = 16 * 1024 * 4 + 16 * 1024 * 2;  // 96KB

    for (int l = 0; l < 4; l++) {
        unsigned short* wqT = wT + l * LW;
        unsigned short* wkT = wqT + (size_t)1024 * 1024;
        unsigned short* wvT = wkT + (size_t)1024 * 1024;
        unsigned short* wcT = wvT + (size_t)1024 * 1024;
        unsigned short* w1T = wcT + (size_t)1024 * 1024;
        unsigned short* w2T = w1T + (size_t)1024 * 4096;

        gemm_bf16_async<<<gP, 256, 0, stream>>>(hbf, wqT, nullptr, qb, nullptr, nullptr, 2048, 1024, 1024, 0);
        gemm_bf16_async<<<gP, 256, 0, stream>>>(hbf, wkT, nullptr, kb, nullptr, nullptr, 2048, 1024, 1024, 0);
        gemm_bf16_async<<<gP, 256, 0, stream>>>(hbf, wvT, nullptr, vb, nullptr, nullptr, 2048, 1024, 1024, 0);
        attention_kernel<<<2048, 32, attn_lds, stream>>>(qb, kb, vb, abf);
        gemm_bf16_async<<<gP, 256, 0, stream>>>(abf, wcT, tb, nullptr, nullptr, hbuf, 2048, 1024, 1024, 0);
        layernorm_kernel<<<2048, 256, 0, stream>>>(tb, g1 + l * 1024, be1 + l * 1024, hbuf, hbf);
        gemm_bf16_async<<<gF, 256, 0, stream>>>(hbf, w1T, nullptr, fbf, b1 + (size_t)l * 4096, nullptr, 2048, 4096, 1024, 1);
        gemm_bf16_async<<<gP, 256, 0, stream>>>(fbf, w2T, tb, nullptr, b2 + l * 1024, hbuf, 2048, 1024, 4096, 0);
        layernorm_kernel<<<2048, 256, 0, stream>>>(tb, g2 + l * 1024, be2 + l * 1024,
                                                   (l == 3) ? out : hbuf,
                                                   (l == 3) ? nullptr : hbf);
    }
}
